// CenterContrastiveLoss_9285719294140
// MI455X (gfx1250) — matrix operand order repaired and checked by reading, not compiled, not run on hardware
//
#include <hip/hip_runtime.h>
#include <hip/hip_bf16.h>

typedef __attribute__((ext_vector_type(2)))  float    v2f;
typedef __attribute__((ext_vector_type(8)))  float    v8f;
typedef __attribute__((ext_vector_type(8)))  _Float16 v8h;
typedef __attribute__((ext_vector_type(16))) _Float16 v16h;

#define EMB   512
#define NCLS  4096
#define BATCH 16384

// ---------------------------------------------------------------------------
// Split centers into f16 hi/lo parts:  c = hi + lo,  |lo| <~ 2^-11 |c|
// ---------------------------------------------------------------------------
__global__ __launch_bounds__(256) void split_centers_kernel(
    const float* __restrict__ centers,
    _Float16* __restrict__ chi, _Float16* __restrict__ clo) {
    const int i = blockIdx.x * 256 + threadIdx.x;   // over NCLS*EMB
    const float x = centers[i];
    const _Float16 h = (_Float16)x;
    chi[i] = h;
    clo[i] = (_Float16)(x - (float)h);
}

// ---------------------------------------------------------------------------
// c_sq[j] = ||centers[j]||^2   (one wave32 per class row, f32)
// ---------------------------------------------------------------------------
__global__ __launch_bounds__(256) void csq_kernel(const float* __restrict__ centers,
                                                  float* __restrict__ csq) {
    const int row  = blockIdx.x * 8 + (threadIdx.x >> 5);
    const int lane = threadIdx.x & 31;
    const float* p = centers + (size_t)row * EMB;
    float s = 0.f;
    for (int k = lane; k < EMB; k += 32) { float v = p[k]; s += v * v; }
    for (int off = 16; off; off >>= 1) s += __shfl_xor(s, off, 32);
    if (lane == 0) csq[row] = s;
}

// ---------------------------------------------------------------------------
// pos[i] = ||e_i - c_{label_i}||^2   (one wave32 per batch row, f32)
// ---------------------------------------------------------------------------
__global__ __launch_bounds__(256) void pos_kernel(const float* __restrict__ emb,
                                                  const int*   __restrict__ labels,
                                                  const float* __restrict__ centers,
                                                  float* __restrict__ pos) {
    const int row  = blockIdx.x * 8 + (threadIdx.x >> 5);
    const int lane = threadIdx.x & 31;
    const int lab  = labels[row];
    const float* e = emb     + (size_t)row * EMB;
    const float* c = centers + (size_t)lab * EMB;
    float s = 0.f;
    for (int k = lane; k < EMB; k += 32) { float d = e[k] - c[k]; s += d * d; }
    for (int off = 16; off; off >>= 1) s += __shfl_xor(s, off, 32);
    if (lane == 0) pos[row] = s;
}

// ---------------------------------------------------------------------------
// FAST PATH: fused GEMM via f16x3 split-precision WMMA + masked row-min.
// Block: 128 threads = 4 wave32s; 16-row embedding tile (hi/lo f16) in LDS.
// Per 16x16 tile: K=512 -> 16 k-chunks x 3 V_WMMA_F32_16X16X32_F16 = 48 WMMA
// (vs 128 for the f32 K=4 shape: 2.7x fewer matrix-pipe issue slots).
// ---------------------------------------------------------------------------
__global__ __launch_bounds__(128) void fused_dist_min_f16x3(
    const float*    __restrict__ emb,
    const int*      __restrict__ labels,
    const _Float16* __restrict__ chi,
    const _Float16* __restrict__ clo,
    const float*    __restrict__ csq,
    float*          __restrict__ negmin)
{
    constexpr int ROWS = 16;
    constexpr int LSTR = 528;                   // f16 row stride; 264 dwords, %64 = 8
    __shared__ __align__(16) _Float16 Ah[ROWS * LSTR];
    __shared__ __align__(16) _Float16 Al[ROWS * LSTR];
    __shared__ float esqp[ROWS][8];
    __shared__ float esq[ROWS];
    __shared__ float minbuf[4][ROWS];

    const int tid     = threadIdx.x;
    const int wave    = tid >> 5;
    const int lane    = tid & 31;
    const int hf      = lane >> 4;
    const int l15     = lane & 15;
    const int rowbase = blockIdx.x * ROWS;

    // Stage A tile: load f32, split into hi/lo f16 LDS tiles.
    for (int i = tid; i < ROWS * (EMB / 4); i += 128) {
        const int r  = i / (EMB / 4);
        const int k4 = (i % (EMB / 4)) * 4;
        const float4 v = ((const float4*)(emb + (size_t)(rowbase + r) * EMB))[k4 >> 2];
        const float x[4] = {v.x, v.y, v.z, v.w};
        #pragma unroll
        for (int j = 0; j < 4; ++j) {
            const _Float16 h = (_Float16)x[j];
            Ah[r * LSTR + k4 + j] = h;
            Al[r * LSTR + k4 + j] = (_Float16)(x[j] - (float)h);
        }
    }
    // Row norms in full f32 from global (L1-hot after staging); 8 threads/row.
    {
        const int r = tid & 15, seg = tid >> 4;
        const float* p = emb + (size_t)(rowbase + r) * EMB + seg * 64;
        float s = 0.f;
        for (int k = 0; k < 64; ++k) s += p[k] * p[k];
        esqp[r][seg] = s;
    }
    __syncthreads();
    if (tid < ROWS) {
        float s = 0.f;
        for (int j = 0; j < 8; ++j) s += esqp[tid][j];
        esq[tid] = s;
    }
    __syncthreads();

    // C layout: slot r -> row (r + 8*hf), col (n0 + l15).
    float esql[8];
    int   labl[8];
    for (int r = 0; r < 8; ++r) {
        const int rl = r + hf * 8;
        esql[r] = esq[rl];
        labl[r] = labels[rowbase + rl];
    }
    float minv[8];
    for (int r = 0; r < 8; ++r) minv[r] = 3.4e38f;

    // A (16x32 f16): lane -> row l15, K = hf*8+[0..7] and 16+hf*8+[0..7].
    const _Float16* ahp = &Ah[l15 * LSTR + hf * 8];
    const _Float16* alp = &Al[l15 * LSTR + hf * 8];

    for (int nt = wave; nt < NCLS / 16; nt += 4) {
        const int col = nt * 16 + l15;
        // B (32x16 f16): lane -> col l15, K = hf*8+[0..7] and 16+hf*8+[0..7].
        const _Float16* bhp = chi + (size_t)col * EMB + hf * 8;
        const _Float16* blp = clo + (size_t)col * EMB + hf * 8;
        if (nt + 4 < NCLS / 16) {                 // warm L1 for this wave's next tile
            __builtin_prefetch(bhp + 64 * EMB, 0, 0);
            __builtin_prefetch(blp + 64 * EMB, 0, 0);
        }

        v8f acc1 = {0.f,0.f,0.f,0.f,0.f,0.f,0.f,0.f};   // hi*hi
        v8f acc2 = {0.f,0.f,0.f,0.f,0.f,0.f,0.f,0.f};   // hi*lo + lo*hi
        #pragma unroll 2
        for (int k0 = 0; k0 < EMB; k0 += 32) {
            v8h ah0 = *(const v8h*)(ahp + k0);
            v8h ah1 = *(const v8h*)(ahp + k0 + 16);
            v8h al0 = *(const v8h*)(alp + k0);
            v8h al1 = *(const v8h*)(alp + k0 + 16);
            v8h bh0 = *(const v8h*)(bhp + k0);
            v8h bh1 = *(const v8h*)(bhp + k0 + 16);
            v8h bl0 = *(const v8h*)(blp + k0);
            v8h bl1 = *(const v8h*)(blp + k0 + 16);
            v16h ah = __builtin_shufflevector(ah0, ah1, 0,1,2,3,4,5,6,7,8,9,10,11,12,13,14,15);
            v16h al = __builtin_shufflevector(al0, al1, 0,1,2,3,4,5,6,7,8,9,10,11,12,13,14,15);
            v16h bh = __builtin_shufflevector(bh0, bh1, 0,1,2,3,4,5,6,7,8,9,10,11,12,13,14,15);
            v16h bl = __builtin_shufflevector(bl0, bl1, 0,1,2,3,4,5,6,7,8,9,10,11,12,13,14,15);
            acc2 = __builtin_amdgcn_wmma_f32_16x16x32_f16(
                false, ah, false, bl, (short)0, acc2, false, false);
            acc2 = __builtin_amdgcn_wmma_f32_16x16x32_f16(
                false, al, false, bh, (short)0, acc2, false, false);
            acc1 = __builtin_amdgcn_wmma_f32_16x16x32_f16(
                false, ah, false, bh, (short)0, acc1, false, false);
        }

        const float cs = csq[col];
        for (int r = 0; r < 8; ++r) {
            float v = esql[r] + cs - 2.0f * (acc1[r] + acc2[r]);
            v = (col == labl[r]) ? 0.0f : v;     // faithful (1 - one_hot) masking
            minv[r] = fminf(minv[r], v);
        }
    }

    // Min across the 16 lanes sharing a row (xor over lane bits 0..3).
    for (int off = 1; off < 16; off <<= 1)
        for (int r = 0; r < 8; ++r)
            minv[r] = fminf(minv[r], __shfl_xor(minv[r], off, 32));

    if (l15 == 0)
        for (int r = 0; r < 8; ++r)
            minbuf[wave][r + hf * 8] = minv[r];
    __syncthreads();

    if (tid < ROWS) {
        const float m = fminf(fminf(minbuf[0][tid], minbuf[1][tid]),
                              fminf(minbuf[2][tid], minbuf[3][tid]));
        negmin[rowbase + tid] = m;
    }
}

// ---------------------------------------------------------------------------
// FALLBACK: full-f32 WMMA path (used only if d_ws is too small for f16 split).
// ---------------------------------------------------------------------------
__global__ __launch_bounds__(128) void fused_dist_min_f32(
    const float* __restrict__ emb,
    const int*   __restrict__ labels,
    const float* __restrict__ centers,
    const float* __restrict__ csq,
    float*       __restrict__ negmin)
{
    constexpr int ROWS = 16;
    constexpr int LSTR = 520;
    __shared__ float Als[ROWS * LSTR];
    __shared__ float esqp[ROWS][8];
    __shared__ float esq[ROWS];
    __shared__ float minbuf[4][ROWS];

    const int tid     = threadIdx.x;
    const int wave    = tid >> 5;
    const int lane    = tid & 31;
    const int hf      = lane >> 4;
    const int l15     = lane & 15;
    const int rowbase = blockIdx.x * ROWS;

    for (int i = tid; i < ROWS * (EMB / 4); i += 128) {
        const int r  = i / (EMB / 4);
        const int k4 = i % (EMB / 4);
        const float4 v = ((const float4*)(emb + (size_t)(rowbase + r) * EMB))[k4];
        float* dst = &Als[r * LSTR + k4 * 4];
        dst[0] = v.x; dst[1] = v.y; dst[2] = v.z; dst[3] = v.w;
    }
    __syncthreads();
    {
        const int r = tid & 15, seg = tid >> 4;
        float s = 0.f;
        const float* p = &Als[r * LSTR + seg * 64];
        for (int k = 0; k < 64; ++k) s += p[k] * p[k];
        esqp[r][seg] = s;
    }
    __syncthreads();
    if (tid < ROWS) {
        float s = 0.f;
        for (int j = 0; j < 8; ++j) s += esqp[tid][j];
        esq[tid] = s;
    }
    __syncthreads();

    float esql[8];
    int   labl[8];
    for (int r = 0; r < 8; ++r) {
        const int rl = r + hf * 8;
        esql[r] = esq[rl];
        labl[r] = labels[rowbase + rl];
    }
    float minv[8];
    for (int r = 0; r < 8; ++r) minv[r] = 3.4e38f;

    const float* aptr = &Als[l15 * LSTR + hf * 2];
    for (int nt = wave; nt < NCLS / 16; nt += 4) {
        const int col = nt * 16 + l15;
        const float* bptr = centers + (size_t)col * EMB + hf * 2;
        v8f acc = {0.f,0.f,0.f,0.f,0.f,0.f,0.f,0.f};
        #pragma unroll 8
        for (int k0 = 0; k0 < EMB; k0 += 4) {
            v2f a = *(const v2f*)(aptr + k0);
            v2f b = *(const v2f*)(bptr + k0);
            acc = __builtin_amdgcn_wmma_f32_16x16x4_f32(
                false, a, false, b, (short)0, acc, false, false);
        }
        const float cs = csq[col];
        for (int r = 0; r < 8; ++r) {
            float v = esql[r] + cs - 2.0f * acc[r];
            v = (col == labl[r]) ? 0.0f : v;
            minv[r] = fminf(minv[r], v);
        }
    }

    for (int off = 1; off < 16; off <<= 1)
        for (int r = 0; r < 8; ++r)
            minv[r] = fminf(minv[r], __shfl_xor(minv[r], off, 32));
    if (l15 == 0)
        for (int r = 0; r < 8; ++r)
            minbuf[wave][r + hf * 8] = minv[r];
    __syncthreads();
    if (tid < ROWS) {
        const float m = fminf(fminf(minbuf[0][tid], minbuf[1][tid]),
                              fminf(minbuf[2][tid], minbuf[3][tid]));
        negmin[rowbase + tid] = m;
    }
}

// ---------------------------------------------------------------------------
// Deterministic final reduction: mean(pos) + mean(negmin) -> out[0]
// ---------------------------------------------------------------------------
__global__ __launch_bounds__(256) void reduce_kernel(const float* __restrict__ pos,
                                                     const float* __restrict__ negmin,
                                                     float* __restrict__ out) {
    __shared__ float sp[256], sn[256];
    float a = 0.f, b = 0.f;
    for (int i = threadIdx.x; i < BATCH; i += 256) { a += pos[i]; b += negmin[i]; }
    sp[threadIdx.x] = a; sn[threadIdx.x] = b;
    __syncthreads();
    for (int s = 128; s; s >>= 1) {
        if ((int)threadIdx.x < s) {
            sp[threadIdx.x] += sp[threadIdx.x + s];
            sn[threadIdx.x] += sn[threadIdx.x + s];
        }
        __syncthreads();
    }
    if (threadIdx.x == 0)
        out[0] = sp[0] / (float)BATCH + sn[0] / (float)BATCH;
}

// ---------------------------------------------------------------------------
extern "C" void kernel_launch(void* const* d_in, const int* in_sizes, int n_in,
                              void* d_out, int out_size, void* d_ws, size_t ws_size,
                              hipStream_t stream) {
    const float* emb     = (const float*)d_in[0];
    const int*   labels  = (const int*)  d_in[1];
    const float* centers = (const float*)d_in[2];
    float*       out     = (float*)d_out;

    char* w = (char*)d_ws;
    float* csq    = (float*)w;                         // [4096]   f32
    float* pos    = (float*)(w + 16384);               // [16384]  f32
    float* negmin = (float*)(w + 16384 + 65536);       // [16384]  f32
    const size_t base_bytes = 16384 + 2 * 65536;       // 147456
    const size_t f16_bytes  = base_bytes + (size_t)2 * NCLS * EMB * sizeof(_Float16) * 2;

    csq_kernel<<<NCLS / 8, 256, 0, stream>>>(centers, csq);
    pos_kernel<<<BATCH / 8, 256, 0, stream>>>(emb, labels, centers, pos);

    if (ws_size >= f16_bytes) {
        _Float16* chi = (_Float16*)(w + base_bytes);
        _Float16* clo = chi + (size_t)NCLS * EMB;
        split_centers_kernel<<<(NCLS * EMB) / 256, 256, 0, stream>>>(centers, chi, clo);
        fused_dist_min_f16x3<<<BATCH / 16, 128, 0, stream>>>(emb, labels, chi, clo, csq, negmin);
    } else {
        fused_dist_min_f32<<<BATCH / 16, 128, 0, stream>>>(emb, labels, centers, csq, negmin);
    }
    reduce_kernel<<<1, 256, 0, stream>>>(pos, negmin, out);
}
